// RingConv2dSimple_19353122636434
// MI455X (gfx1250) — matrix operand order repaired and checked by reading, not compiled, not run on hardware
//
#include <hip/hip_runtime.h>
#include <hip/hip_bf16.h>
#include <math.h>

typedef _Float16 h8  __attribute__((ext_vector_type(8)));
typedef _Float16 h16 __attribute__((ext_vector_type(16)));
typedef float    f8  __attribute__((ext_vector_type(8)));

#if defined(__has_builtin) && __has_builtin(__builtin_amdgcn_sched_barrier)
#define SCHED_FENCE() __builtin_amdgcn_sched_barrier(0)
#else
#define SCHED_FENCE()
#endif

#define CIN    64
#define COUT   128
#define HH     128
#define WW     128
#define BATCH  16
#define TILE_W 64
#define COLS   66      // TILE_W + 2 halo columns

// ---------------------------------------------------------------------------
// Weight repack: f32 [Cin][Cout][3][3]  ->  f16 [khw][Cout][Cin] in d_ws.
// Makes the WMMA A-fragment (16 couts x 32 cins) two contiguous b128 loads.
// ---------------------------------------------------------------------------
__global__ __launch_bounds__(256) void ring_w2h(const float* __restrict__ w,
                                                _Float16* __restrict__ wf) {
  int i = blockIdx.x * 256 + threadIdx.x;   // i in [0, 9*128*64)
  if (i < 9 * COUT * CIN) {
    int cin  = i & (CIN - 1);
    int cout = (i >> 6) & (COUT - 1);
    int khw  = i >> 13;                      // 0..8
    float v = w[(cin * COUT + cout) * 9 + khw];
    wf[(khw * COUT + cout) * CIN + cin] = (_Float16)v;
  }
}

// ---------------------------------------------------------------------------
// Main kernel: one workgroup = one (b, y) row, 64-px tile, all 128 Cout.
// 8 waves; wave w computes couts [16w, 16w+16) for all 64 px, with two f32
// accumulators per 16x16 tile (dir_x from cos, dir_y from sin).
// Inner product: 18 k-steps x 8 WMMA = 144 WMMAs, software-pipelined with
// double-buffered A (global) / B (LDS) fragments; sched_barriers keep the
// prefetch group ahead of the consuming WMMA group.
// ---------------------------------------------------------------------------
__global__ __launch_bounds__(256, 2) void ring_conv(const float* __restrict__ x,
                                                    const _Float16* __restrict__ wf,
                                                    float* __restrict__ out) {
  // cos/sin tiles, cin innermost so B-fragments are contiguous ds_load_b128s
  __shared__ _Float16 sC[3 * COLS * CIN];   // 25,344 B
  __shared__ _Float16 sS[3 * COLS * CIN];   // 25,344 B

  const int bid = blockIdx.x;               // 0 .. 16*128*2-1
  const int b   = bid >> 8;                 // 128 rows * 2 tiles per batch
  const int r   = bid & 255;
  const int y   = r >> 1;
  const int x0  = (r & 1) * TILE_W;

  // ---- stage cos/sin of the 3-row halo into LDS (zero-pad BEFORE cos/sin) --
  const int TOT = 3 * COLS * CIN;           // 12,672 elements
  for (int idx = threadIdx.x; idx < TOT; idx += 256) {
    int c   = idx % COLS;                   // halo column 0..65 (coalesced x reads)
    int q   = idx / COLS;
    int cin = q & (CIN - 1);
    int dy  = q >> 6;                       // 0..2
    int row = y - 1 + dy;
    int col = x0 - 1 + c;
    float v = 0.0f;
    if ((unsigned)row < (unsigned)HH && (unsigned)col < (unsigned)WW)
      v = x[((b * CIN + cin) * HH + row) * WW + col];
    int so = (dy * COLS + c) * CIN + cin;
    sC[so] = (_Float16)__cosf(v);           // pad cell -> cos(0)=1
    sS[so] = (_Float16)__sinf(v);           // pad cell -> sin(0)=0
  }
  __syncthreads();

  const int lane     = threadIdx.x & 31;
  const int wave     = threadIdx.x >> 5;
  const int m        = lane & 15;           // A row / B col / D col index
  const int grp      = lane >> 4;           // half-wave group
  const int coutBase = wave * 16;

  f8 accX[4] = {{0,0,0,0,0,0,0,0},{0,0,0,0,0,0,0,0},
                {0,0,0,0,0,0,0,0},{0,0,0,0,0,0,0,0}};
  f8 accY[4] = {{0,0,0,0,0,0,0,0},{0,0,0,0,0,0,0,0},
                {0,0,0,0,0,0,0,0},{0,0,0,0,0,0,0,0}};

  // A fragment for k-step k (khw = k>>1, cin chunk = k&1):
  // 16x32 f16; lane halves 0..15 -> K=grp*16+h
  auto loadA = [&](int k) -> h16 {
    const int khw = k >> 1, cc = k & 1;
    const _Float16* wp = wf + ((khw * COUT) + coutBase + m) * CIN + cc * 32;
    h8 a0 = *(const h8*)(wp + grp * 16);
    h8 a1 = *(const h8*)(wp + grp * 16 + 8);
    return __builtin_shufflevector(a0, a1, 0,1,2,3,4,5,6,7,8,9,10,11,12,13,14,15);
  };

  // B fragments (32 cin x 16 px) for all 4 pixel sub-tiles, cos + sin.
  auto loadB = [&](int k, h16 (&Bc)[4], h16 (&Bs)[4]) {
    const int khw = k >> 1, cc = k & 1;
    const int kh = khw / 3, kw = khw % 3;
#pragma unroll
    for (int t = 0; t < 4; ++t) {
      int col = t * 16 + m + kw;            // halo column for this output px
      int so  = (kh * COLS + col) * CIN + cc * 32 + grp * 16;   // 32B aligned
      h8 b0 = *(const h8*)(&sC[so]);
      h8 b1 = *(const h8*)(&sC[so + 8]);
      Bc[t] = __builtin_shufflevector(b0, b1, 0,1,2,3,4,5,6,7,8,9,10,11,12,13,14,15);
      h8 s0 = *(const h8*)(&sS[so]);
      h8 s1 = *(const h8*)(&sS[so + 8]);
      Bs[t] = __builtin_shufflevector(s0, s1, 0,1,2,3,4,5,6,7,8,9,10,11,12,13,14,15);
    }
  };

  // 8 back-to-back WMMAs: distinct accumulators, shared A -> no hazards.
  auto wmma8 = [&](h16 A, h16 (&Bc)[4], h16 (&Bs)[4]) {
#pragma unroll
    for (int t = 0; t < 4; ++t) {
      accX[t] = __builtin_amdgcn_wmma_f32_16x16x32_f16(
          false, A, false, Bc[t], (short)0, accX[t], false, false);
      accY[t] = __builtin_amdgcn_wmma_f32_16x16x32_f16(
          false, A, false, Bs[t], (short)0, accY[t], false, false);
    }
  };

  // ---- software-pipelined main loop: prefetch k+1 before consuming k -------
  // sched_barrier(0) fences keep each prefetch group ahead of the WMMA group
  // so the allocator must keep both buffers live (no ds_load->wait0->wmma).
  h16 B0c[4], B0s[4], B1c[4], B1s[4];
  h16 A0 = loadA(0), A1;
  loadB(0, B0c, B0s);
#pragma unroll
  for (int k = 0; k < 18; k += 2) {
    A1 = loadA(k + 1);
    loadB(k + 1, B1c, B1s);
    SCHED_FENCE();
    wmma8(A0, B0c, B0s);                    // waits only on step-k loads
    SCHED_FENCE();
    if (k + 2 < 18) {
      A0 = loadA(k + 2);
      loadB(k + 2, B0c, B0s);
    }
    SCHED_FENCE();
    wmma8(A1, B1c, B1s);                    // waits only on step-(k+1) loads
    SCHED_FENCE();
  }

  // ---- epilogue: atan2(dir_y, dir_x) -> out[b][cout][y][px] ----------------
#pragma unroll
  for (int t = 0; t < 4; ++t) {
#pragma unroll
    for (int v = 0; v < 8; ++v) {
      int cout = coutBase + grp * 8 + v;    // C/D layout: lanes 16-31 hold M+8
      int px   = x0 + t * 16 + m;
      out[((b * COUT + cout) * HH + y) * WW + px] = atan2f(accY[t][v], accX[t][v]);
    }
  }
}

// ---------------------------------------------------------------------------
extern "C" void kernel_launch(void* const* d_in, const int* in_sizes, int n_in,
                              void* d_out, int out_size, void* d_ws, size_t ws_size,
                              hipStream_t stream) {
  (void)in_sizes; (void)n_in; (void)out_size; (void)ws_size;
  const float* x = (const float*)d_in[0];
  const float* w = (const float*)d_in[1];
  // d_in[2] = stride(1), d_in[3] = padding(1): baked into the kernel.
  _Float16* wf16 = (_Float16*)d_ws;         // needs 9*128*64*2 = 147,456 B

  ring_w2h<<<(9 * COUT * CIN + 255) / 256, 256, 0, stream>>>(w, wf16);
  ring_conv<<<BATCH * HH * (WW / TILE_W), 256, 0, stream>>>(x, wf16,
                                                            (float*)d_out);
}
